// Spatial_dot_80951543595332
// MI455X (gfx1250) — matrix operand order repaired and checked by reading, not compiled, not run on hardware
//
#include <hip/hip_runtime.h>

typedef _Float16 half_t;
typedef __attribute__((ext_vector_type(16))) _Float16 v16h;
typedef __attribute__((ext_vector_type(8)))  float    v8f;

#define T_ 8
#define C_ 128
#define N_ 4096
#define F_ 64

#define MCHUNK   32      // K/G rows staged per iteration
#define BLK_ROWS 128     // Q rows per workgroup (4 waves x 32)

// ---------------------------------------------------------------------------
// Projection: acc[f] = sum_c W[f][c] * x[t][c][n] + b[f]
//   TRANSPOSED=0 -> dst[t][n][f]   (Q, K layout)
//   TRANSPOSED=1 -> dst[t][f][n]   (G layout: contiguous async staging)
// ---------------------------------------------------------------------------
template <int TRANSPOSED>
__global__ __launch_bounds__(256)
void proj_kernel(const float* __restrict__ x, const float* __restrict__ Wm,
                 const float* __restrict__ bias, half_t* __restrict__ dst) {
    const int n = blockIdx.x * 256 + threadIdx.x;
    const int t = blockIdx.y;

    float acc[F_];
#pragma unroll
    for (int f = 0; f < F_; ++f) acc[f] = 0.0f;

    const float* xp = x + (size_t)t * C_ * N_ + n;
    for (int c = 0; c < C_; ++c) {
        const float xv = xp[(size_t)c * N_];
        const float* wc = Wm + c;
#pragma unroll
        for (int f = 0; f < F_; ++f) acc[f] = fmaf(wc[(size_t)f * C_], xv, acc[f]);
    }

    if (TRANSPOSED) {
        half_t* dp = dst + (size_t)t * F_ * N_ + n;
#pragma unroll
        for (int f = 0; f < F_; ++f) dp[(size_t)f * N_] = (half_t)(acc[f] + bias[f]);
    } else {
        unsigned int* dp32 = (unsigned int*)(dst + ((size_t)t * N_ + n) * F_);
#pragma unroll
        for (int j = 0; j < F_ / 2; ++j) {
            union { unsigned int u; half_t h[2]; } p;
            p.h[0] = (half_t)(acc[2 * j]     + bias[2 * j]);
            p.h[1] = (half_t)(acc[2 * j + 1] + bias[2 * j + 1]);
            dp32[j] = p.u;
        }
    }
}

// ---------------------------------------------------------------------------
// Async global -> LDS copy of 16 bytes per lane (CDNA5, ASYNCcnt-tracked)
// ---------------------------------------------------------------------------
__device__ __forceinline__ void async_cp16(unsigned lds_off, const void* gaddr) {
    asm volatile("global_load_async_to_lds_b128 %0, %1, off"
                 :: "v"(lds_off), "v"(gaddr) : "memory");
}

union AOp { v16h v; unsigned int u[8]; };

// ---------------------------------------------------------------------------
// Attention: out[t][f][n] = sum_m relu(q[n]·k[m]) * g[m][f]
// 4 waves/block, each wave owns 32 Q rows. Double-buffered async LDS staging,
// single barrier per chunk (issue happens after the barrier).
// ---------------------------------------------------------------------------
__global__ __launch_bounds__(128)
void attn_kernel(const half_t* __restrict__ qb, const half_t* __restrict__ kb,
                 const half_t* __restrict__ gbt, float* __restrict__ out) {
    __shared__ __align__(16) half_t kt[2][MCHUNK][F_];     // K chunk, [m][f]
    __shared__ __align__(16) half_t gt[2][F_][MCHUNK];     // G chunk, [f][m]
    __shared__ __align__(16) half_t st[4][32][MCHUNK];     // per-wave relu(S) scratch

    const int tid  = threadIdx.x;
    const int lane = tid & 31;
    const int wave = tid >> 5;
    const int t    = blockIdx.y;
    const int row_base = blockIdx.x * BLK_ROWS + wave * 32;

    const int lrow = lane & 15;
    const int lhi  = lane >> 4;

    // --- Q A-operands: 2 row tiles x 2 f-chunks, resident all loop ---
    AOp qa[2][2];
#pragma unroll
    for (int rt = 0; rt < 2; ++rt) {
        const unsigned int* qrow32 =
            (const unsigned int*)(qb + ((size_t)t * N_ + (row_base + rt * 16 + lrow)) * F_);
#pragma unroll
        for (int fc = 0; fc < 2; ++fc)
#pragma unroll
            for (int j = 0; j < 8; ++j) {
                const int k = 32 * fc + 16 * (j >> 2) + 2 * (j & 3) + 8 * lhi;
                qa[rt][fc].u[j] = qrow32[k >> 1];
            }
    }

    v8f o[2][4] = {};   // O accumulators: 2 row tiles x 4 f tiles

    // --- per-thread async staging addresses ---
    const half_t* kgl = kb + (size_t)t * N_ * F_ + (size_t)(tid >> 2) * F_ + (tid & 3) * 16;
    const half_t* ggl = gbt + ((size_t)t * F_ + (tid >> 1)) * N_ + (tid & 1) * 16;
    unsigned kld[2], gld[2];
#pragma unroll
    for (int b = 0; b < 2; ++b) {
        kld[b] = (unsigned)(uintptr_t)&kt[b][tid >> 2][(tid & 3) * 16];
        gld[b] = (unsigned)(uintptr_t)&gt[b][tid >> 1][(tid & 1) * 16];
    }

    // prologue: stage chunk 0 into buffer 0
    async_cp16(kld[0],      kgl);
    async_cp16(kld[0] + 16, kgl + 8);
    async_cp16(gld[0],      ggl);
    async_cp16(gld[0] + 16, ggl + 8);

    for (int i = 0; i < N_ / MCHUNK; ++i) {
        const int m0  = i * MCHUNK;
        const int cur = i & 1;

        asm volatile("s_wait_asynccnt 0x0" ::: "memory");  // own copies of chunk i landed
        __syncthreads();  // tile resident everywhere; buf[cur^1] readers (iter i-1) done

        if (m0 + MCHUNK < N_) {        // uniform: EXEC stays full for WMMA
            const half_t* ks = kgl + (size_t)(m0 + MCHUNK) * F_;
            const half_t* gs = ggl + (m0 + MCHUNK);
            async_cp16(kld[cur ^ 1],      ks);
            async_cp16(kld[cur ^ 1] + 16, ks + 8);
            async_cp16(gld[cur ^ 1],      gs);
            async_cp16(gld[cur ^ 1] + 16, gs + 8);
        }

        // --- GEMM1: S(32x32) = Q(32x64) · K^T; B operands reused over row tiles ---
        v8f s[2][2];
#pragma unroll
        for (int mt = 0; mt < 2; ++mt) {
            AOp bk[2];
#pragma unroll
            for (int fc = 0; fc < 2; ++fc)
#pragma unroll
                for (int j = 0; j < 8; ++j) {
                    const int kk = 32 * fc + 16 * (j >> 2) + 2 * (j & 3) + 8 * lhi;     // f index
                    bk[fc].u[j] = *(const unsigned int*)&kt[cur][mt * 16 + lrow][kk];
                }
#pragma unroll
            for (int rt = 0; rt < 2; ++rt) {
                v8f c = {};
                c = __builtin_amdgcn_wmma_f32_16x16x32_f16(false, qa[rt][0].v, false, bk[0].v,
                                                           (short)0, c, false, false);
                c = __builtin_amdgcn_wmma_f32_16x16x32_f16(false, qa[rt][1].v, false, bk[1].v,
                                                           (short)0, c, false, false);
                s[rt][mt] = c;
            }
        }

        // --- relu + f32->f16, C/D -> A re-layout via wave-private LDS ---
#pragma unroll
        for (int rt = 0; rt < 2; ++rt)
#pragma unroll
            for (int mt = 0; mt < 2; ++mt)
#pragma unroll
                for (int r = 0; r < 8; ++r) {
                    const float v = fmaxf(s[rt][mt][r], 0.0f);
                    st[wave][rt * 16 + r + 8 * lhi][mt * 16 + lrow] = (half_t)v;
                }
        AOp as[2];
#pragma unroll
        for (int rt = 0; rt < 2; ++rt)
#pragma unroll
            for (int j = 0; j < 8; ++j) {
                const int k = 16 * (j >> 2) + 2 * (j & 3) + 8 * lhi;   // m_local
                as[rt].u[j] = *(const unsigned int*)&st[wave][rt * 16 + lrow][k];
            }

        // --- GEMM2: O(32x64) += relu(S)(32x32) · G(32x64) ---
#pragma unroll
        for (int ft = 0; ft < 4; ++ft) {
            AOp bg;
#pragma unroll
            for (int j = 0; j < 8; ++j) {
                const int kk = 16 * (j >> 2) + 2 * (j & 3) + 8 * lhi;                       // m_local
                bg.u[j] = *(const unsigned int*)&gt[cur][ft * 16 + lrow][kk];
            }
#pragma unroll
            for (int rt = 0; rt < 2; ++rt)
                o[rt][ft] = __builtin_amdgcn_wmma_f32_16x16x32_f16(false, as[rt].v, false, bg.v,
                                                                  (short)0, o[rt][ft], false, false);
        }
    }

    // --- write O: out[t][f][n]; C/D layout lane->col, vgpr r->row ---
    float* op = out + (size_t)t * F_ * N_;
#pragma unroll
    for (int rt = 0; rt < 2; ++rt)
#pragma unroll
        for (int ft = 0; ft < 4; ++ft) {
            const int f = ft * 16 + lrow;
#pragma unroll
            for (int r = 0; r < 8; ++r) {
                const int n = row_base + rt * 16 + r + 8 * lhi;
                op[(size_t)f * N_ + n] = o[rt][ft][r];
            }
        }
}

// ---------------------------------------------------------------------------
extern "C" void kernel_launch(void* const* d_in, const int* in_sizes, int n_in,
                              void* d_out, int out_size, void* d_ws, size_t ws_size,
                              hipStream_t stream) {
    (void)in_sizes; (void)n_in; (void)out_size; (void)ws_size;

    const float* input1 = (const float*)d_in[0];
    const float* input2 = (const float*)d_in[1];
    const float* W1 = (const float*)d_in[2];
    const float* b1 = (const float*)d_in[3];
    const float* W2 = (const float*)d_in[4];
    const float* b2 = (const float*)d_in[5];
    const float* W3 = (const float*)d_in[6];
    const float* b3 = (const float*)d_in[7];
    float* out = (float*)d_out;

    half_t* qbuf  = (half_t*)d_ws;                       // [T][N][F]
    half_t* kbuf  = qbuf + (size_t)T_ * N_ * F_;         // [T][N][F]
    half_t* gbuft = kbuf + (size_t)T_ * N_ * F_;         // [T][F][N] (transposed)

    dim3 pgrid(N_ / 256, T_);
    proj_kernel<0><<<pgrid, 256, 0, stream>>>(input1, W1, b1, qbuf);
    proj_kernel<0><<<pgrid, 256, 0, stream>>>(input2, W2, b2, kbuf);
    proj_kernel<1><<<pgrid, 256, 0, stream>>>(input1, W3, b3, gbuft);

    dim3 agrid(N_ / BLK_ROWS, T_);
    attn_kernel<<<agrid, 128, 0, stream>>>(qbuf, kbuf, gbuft, out);
}
